// GIN_77360950936124
// MI455X (gfx1250) — matrix operand order repaired and checked by reading, not compiled, not run on hardware
//
#include <hip/hip_runtime.h>
#include <hip/hip_bf16.h>
#include <math.h>

// ---------------------------------------------------------------------------
// CDNA5 (gfx1250) GINE-GNN pipeline.
//   - node-update GEMMs via v_wmma_f32_16x16x32_bf16 (wave32, 16x16 tile/wave),
//     templated so K-loops fully unroll; B operand staged into LDS by the
//     Tensor Data Mover (tensor_load_to_lds + s_wait_tensorcnt) in a
//     pre-swizzled layout so each lane reads its 16 B elements as 2x
//     ds_load_b128; A prefetched with global_prefetch_b8
//   - edge message passing via gather + global f32 atomics
//   - set2set via small VALU kernels + ordered-uint atomic max + f32 atomics
//   - head via per-graph block with LDS tree reduction
// ---------------------------------------------------------------------------

typedef __attribute__((ext_vector_type(16))) __bf16 v16bf;
typedef __attribute__((ext_vector_type(8)))  float  v8f;
typedef __attribute__((ext_vector_type(4)))  unsigned int u32x4;
typedef __attribute__((ext_vector_type(8)))  int i32x8;
typedef __attribute__((ext_vector_type(4)))  int i32x4;

// ---------------------------- utility kernels ------------------------------

__global__ void zero_f32(float* __restrict__ p, size_t n) {
  size_t i = (size_t)blockIdx.x * blockDim.x + threadIdx.x;
  if (i < n) p[i] = 0.0f;
}

// Build combined GEMM operand:  Wc = [[nn_w + l_w]; [nn_w]]  (2*dIn x dOut)
// stored pre-swizzled into the WMMA B-register layout so that in LDS each
// lane's 16 bf16 values for one v_wmma are 32 contiguous bytes:
//   element (k, j) -> Wc[ (((k>>5)*2 + ((k>>4)&1)) * dOut + j) * 16 + (k&15) ]
// Also bc = nn_b + l_b.
__global__ void combine_weights(const float* __restrict__ nnw,
                                const float* __restrict__ lw,
                                const float* __restrict__ nnb,
                                const float* __restrict__ lb,
                                __bf16* __restrict__ Wc,
                                float* __restrict__ bc,
                                int dIn, int dOut) {
  int idx = blockIdx.x * blockDim.x + threadIdx.x;
  int tot = 2 * dIn * dOut;
  if (idx < tot) {
    int k = idx / dOut;
    int j = idx - k * dOut;
    float v = (k < dIn) ? (nnw[k * dOut + j] + lw[k * dOut + j])
                        : nnw[(k - dIn) * dOut + j];
    int swz = (((k >> 5) * 2 + ((k >> 4) & 1)) * dOut + j) * 16 + (k & 15);
    Wc[swz] = (__bf16)v;
  }
  if (idx < dOut) bc[idx] = nnb[idx] + lb[idx];
}

// -------------------------- edge message kernel ----------------------------
// One thread per (edge, feature): msg = relu(x[src] + edge_attr@ew + eb),
// scatter-add into aggr[dst] with global f32 atomics.
__global__ void msg_aggr(const float* __restrict__ x,
                         const float* __restrict__ ea,    // [E,6]
                         const int*   __restrict__ srcv,
                         const int*   __restrict__ dstv,
                         const float* __restrict__ ew,    // [6,d]
                         const float* __restrict__ eb,    // [d]
                         float* __restrict__ aggr,        // [N,d]
                         int E, int d) {
  size_t tid = (size_t)blockIdx.x * blockDim.x + threadIdx.x;
  int e = (int)(tid / d);
  int j = (int)(tid - (size_t)e * d);
  if (e >= E) return;
  int s = srcv[e];
  int t = dstv[e];
  const float* ear = ea + (size_t)e * 6;
  float ac = eb[j];
#pragma unroll
  for (int k = 0; k < 6; ++k) ac = fmaf(ear[k], ew[k * d + j], ac);
  ac += x[(size_t)s * d + j];
  ac = fmaxf(ac, 0.0f);
  atomicAdd(&aggr[(size_t)t * d + j], ac);
}

// ------------------------------ WMMA GEMM ----------------------------------
// Out[N,DOUT] = [Ax | Ag] (N x 2*DIN, f32 -> bf16) @ Wc (2*DIN x DOUT, bf16) + bc
//
// Block = 256 threads = 8 waves, covers BM=64 rows x all DOUT columns.
// Wc (pre-swizzled) is staged into LDS once per block — by the Tensor Data
// Mover when available (1-D D# descriptor, TENSORcnt ordering), else by a
// cooperative vector copy. Each wave then computes 16x16 output tiles with
// v_wmma_f32_16x16x32_bf16 (f32 accumulate), fully unrolled over K2=2*DIN.
//
// ISA operand layouts (CDNA5 ISA 7.12.2, wave32):
//   A (16x32 bf16): lane L -> row M = L&15; element e -> K = e + 16*(L>>4) (code's k order, not ISA table)
//   B (32x16 bf16): lane L -> col N = L&15; element e -> K = e + 16*(L>>4), same as A (code's order, not ISA table)
//                   (swizzled storage makes these 16 values contiguous)
//   C/D (16x16 f32): lane L -> col N = L&15; vgpr r -> row M = r + 8*(L>>4)
template <int DIN, int DOUT, bool ELU>
__global__ __launch_bounds__(256) void gemm_wmma(
    const float* __restrict__ Ax,   // [N, DIN]
    const float* __restrict__ Ag,   // [N, DIN]
    const __bf16* __restrict__ Wc,  // [2*DIN * DOUT], swizzled
    const float* __restrict__ bc,   // [DOUT]
    float* __restrict__ Out,        // [N, DOUT]
    int Nrows) {
  constexpr int K2 = 2 * DIN;
  constexpr int COL_TILES = DOUT / 16;          // 8 (DOUT=128) or 4 (DOUT=64)
  constexpr int WAVES_PER_COL = 8 / COL_TILES;  // 1 or 2
  constexpr int BM = 64;                        // rows per block
  constexpr int ROW_TILES = BM / 16;            // 4

  __shared__ __align__(32) __bf16 Bs[K2 * DOUT];  // 16..64 KB

#if __has_builtin(__builtin_amdgcn_tensor_load_to_lds)
  // ---- TDM stage: wave 0 issues one 1-D tensor DMA of Wc into LDS ----
  if (threadIdx.x < 32) {
    constexpr unsigned elems = (unsigned)(K2 * DOUT * sizeof(__bf16) / 4);  // 4B units
    unsigned lds = (unsigned)(size_t)(&Bs[0]);
    unsigned long long ga = (unsigned long long)(size_t)Wc;
    u32x4 g0;
    g0[0] = 1u;                                            // count=1, user, no gather
    g0[1] = lds;                                           // lds_addr (bytes)
    g0[2] = (unsigned)(ga & 0xffffffffu);                  // global_addr[31:0]
    g0[3] = (unsigned)((ga >> 32) & 0x01ffffffu)           // global_addr[56:32]
            | (2u << 30);                                  // type = 2 ("image")
    i32x8 g1;
    g1[0] = (int)(2u << 16);                               // data_size = 4 bytes
    g1[1] = (int)((elems & 0xffffu) << 16);                // tensor_dim0[15:0]
    g1[2] = (int)(((elems >> 16) & 0xffffu) | (1u << 16)); // tensor_dim0[31:16], tensor_dim1=1
    g1[3] = (int)((elems & 0xffffu) << 16);                // tile_dim0 = elems
    g1[4] = 0;                                             // tile_dim1/2 unused
    g1[5] = (int)elems;                                    // tensor_dim0_stride
    g1[6] = 0;
    g1[7] = 0;
    i32x4 gz = {0, 0, 0, 0};
#if defined(__clang_major__) && (__clang_major__ >= 23)
    i32x8 gz8 = {0, 0, 0, 0, 0, 0, 0, 0};
    __builtin_amdgcn_tensor_load_to_lds(g0, g1, gz, gz, gz8, 0);
#else
    __builtin_amdgcn_tensor_load_to_lds(g0, g1, gz, gz, 0);
#endif
    __builtin_amdgcn_s_wait_tensorcnt(0);
  }
#else
  // ---- fallback: cooperative vector copy (layout-agnostic) ----
  {
    constexpr int CHUNKS = K2 * DOUT / 8;  // 8 bf16 per uint4
    const uint4* src = (const uint4*)Wc;
    uint4* dst = (uint4*)Bs;
#pragma unroll
    for (int i = threadIdx.x; i < CHUNKS; i += 256) dst[i] = src[i];
  }
#endif
  __syncthreads();

  const int lane = threadIdx.x & 31;
  const int wave = threadIdx.x >> 5;
  const int ln = lane & 15;
  const int kh = lane >> 4;
  const int colTile = wave % COL_TILES;
  const int bcol = colTile * 16 + ln;
  const float bias = bc[bcol];
  const int rowBase = blockIdx.x * BM;
  const v16bf* BsV = (const v16bf*)Bs;  // one v16bf = one lane's B fragment

#pragma unroll
  for (int rs = wave / COL_TILES; rs < ROW_TILES; rs += WAVES_PER_COL) {
    const int row0 = rowBase + rs * 16;
    if (row0 >= Nrows) break;  // wave-uniform guard (EXEC stays all-ones)
    const int arow = row0 + ln;

    // Prefetch the next row-subtile's A rows while we compute this one.
    {
      const int nrow = arow + WAVES_PER_COL * 16;
      if (nrow < Nrows) {
        __builtin_prefetch(Ax + (size_t)nrow * DIN, 0, 0);
        __builtin_prefetch(Ag + (size_t)nrow * DIN, 0, 0);
      }
    }

    v8f acc = {};
#pragma unroll
    for (int kt = 0; kt < K2; kt += 32) {
      // Each 32-wide K chunk lies entirely inside Ax or Ag (DIN % 32 == 0).
      const float* srcA = (kt < DIN) ? (Ax + (size_t)arow * DIN + kt)
                                     : (Ag + (size_t)arow * DIN + (kt - DIN));
      v16bf a;
#pragma unroll
      for (int e = 0; e < 8; ++e) {
        a[e]     = (__bf16)srcA[16 * kh + e];      // K = kt + 16*kh + e
        a[e + 8] = (__bf16)srcA[16 * kh + e + 8];  // K = kt + 16*kh + 8 + e
      }
      // Swizzled B fragment: 32 contiguous bytes -> 2x ds_load_b128.
      v16bf b = BsV[((kt >> 5) * 2 + kh) * DOUT + bcol];
      acc = __builtin_amdgcn_wmma_f32_16x16x32_bf16(
          false, a, false, b, (short)0, acc, false, false);
    }

#pragma unroll
    for (int r = 0; r < 8; ++r) {
      const int m = r + 8 * kh;
      float v = acc[r] + bias;
      if (ELU) v = (v > 0.0f) ? v : (expf(v) - 1.0f);
      Out[(size_t)(row0 + m) * DOUT + bcol] = v;
    }
  }
}

// ------------------------------- Set2Set -----------------------------------

__device__ __forceinline__ float sigf(float x) { return 1.0f / (1.0f + expf(-x)); }

// One thread per (graph, unit): full LSTM cell (torch gate order i,f,g,o).
__global__ void lstm_step(const float* __restrict__ qstar,  // [B,128]
                          const float* __restrict__ hin,    // [B,64]
                          const float* __restrict__ wih,    // [128,256]
                          const float* __restrict__ whh,    // [64,256]
                          const float* __restrict__ bih,    // [256]
                          const float* __restrict__ bhh,    // [256]
                          float* __restrict__ hout,         // [B,64]
                          float* __restrict__ cbuf,         // [B,64] in/out
                          int B) {
  int idx = blockIdx.x * blockDim.x + threadIdx.x;
  if (idx >= B * 64) return;
  int g = idx >> 6;
  int j = idx & 63;
  float gi = bih[j]       + bhh[j];
  float gf = bih[64 + j]  + bhh[64 + j];
  float gg = bih[128 + j] + bhh[128 + j];
  float go = bih[192 + j] + bhh[192 + j];
  const float* q = qstar + (size_t)g * 128;
  for (int k = 0; k < 128; ++k) {
    float qv = q[k];
    const float* wr = wih + (size_t)k * 256;
    gi = fmaf(qv, wr[j], gi);
    gf = fmaf(qv, wr[64 + j], gf);
    gg = fmaf(qv, wr[128 + j], gg);
    go = fmaf(qv, wr[192 + j], go);
  }
  const float* h = hin + (size_t)g * 64;
  for (int k = 0; k < 64; ++k) {
    float hv = h[k];
    const float* wr = whh + (size_t)k * 256;
    gi = fmaf(hv, wr[j], gi);
    gf = fmaf(hv, wr[64 + j], gf);
    gg = fmaf(hv, wr[128 + j], gg);
    go = fmaf(hv, wr[192 + j], go);
  }
  float cn = sigf(gf) * cbuf[idx] + sigf(gi) * tanhf(gg);
  cbuf[idx] = cn;
  hout[idx] = sigf(go) * tanhf(cn);
}

// e[n] = dot(h3[n], q[batch[n]]); segment max via ordered-uint atomicMax.
__global__ void attn_score(const float* __restrict__ h3,   // [N,64]
                           const float* __restrict__ hq,   // [B,64]
                           const int* __restrict__ batch,
                           float* __restrict__ evec,       // [N]
                           unsigned* __restrict__ emax,    // [B] (zero-initialized)
                           int N) {
  int n = blockIdx.x * blockDim.x + threadIdx.x;
  if (n >= N) return;
  int g = batch[n];
  const float* xr = h3 + (size_t)n * 64;
  const float* qr = hq + (size_t)g * 64;
  float s = 0.0f;
#pragma unroll 8
  for (int j = 0; j < 64; ++j) s = fmaf(xr[j], qr[j], s);
  evec[n] = s;
  unsigned u = __float_as_uint(s);
  unsigned key = (u & 0x80000000u) ? ~u : (u | 0x80000000u);
  atomicMax(&emax[g], key);
}

__global__ void attn_expsum(const float* __restrict__ evec,
                            const unsigned* __restrict__ emax,
                            const int* __restrict__ batch,
                            float* __restrict__ avec,  // [N]
                            float* __restrict__ asum,  // [B]
                            int N) {
  int n = blockIdx.x * blockDim.x + threadIdx.x;
  if (n >= N) return;
  int g = batch[n];
  unsigned k = emax[g];
  unsigned u = (k & 0x80000000u) ? (k & 0x7fffffffu) : ~k;
  float m = __uint_as_float(u);
  float a = expf(evec[n] - m);
  avec[n] = a;
  atomicAdd(&asum[g], a);
}

// r[g] += (a[n]/asum[g]) * h3[n]
__global__ void attn_readout(const float* __restrict__ h3,
                             const float* __restrict__ avec,
                             const float* __restrict__ asum,
                             const int* __restrict__ batch,
                             float* __restrict__ r,  // [B,64]
                             int N) {
  size_t idx = (size_t)blockIdx.x * blockDim.x + threadIdx.x;
  if (idx >= (size_t)N * 64) return;
  int n = (int)(idx >> 6);
  int j = (int)(idx & 63);
  int g = batch[n];
  atomicAdd(&r[(size_t)g * 64 + j], (avec[n] / asum[g]) * h3[idx]);
}

__global__ void qstar_update(const float* __restrict__ hnew,  // [B,64]
                             const float* __restrict__ r,     // [B,64]
                             float* __restrict__ qstar,       // [B,128]
                             int B) {
  int idx = blockIdx.x * blockDim.x + threadIdx.x;
  if (idx >= B * 128) return;
  int g = idx >> 7;
  int j = idx & 127;
  qstar[idx] = (j < 64) ? hnew[(size_t)g * 64 + j] : r[(size_t)g * 64 + (j - 64)];
}

// ------------------------------- MLP head ----------------------------------
// One block (128 threads) per graph: relu([q_star,t,p] @ l4 + b4) @ l5 + b5,
// final dot via LDS tree reduction.
__global__ __launch_bounds__(128) void head_kernel(
    const float* __restrict__ qstar,  // [B,128]
    const float* __restrict__ tvec,   // [B]
    const float* __restrict__ pvec,   // [B]
    const float* __restrict__ w4,     // [130,128]
    const float* __restrict__ b4,     // [128]
    const float* __restrict__ w5,     // [128]
    const float* __restrict__ b5,     // [1]
    float* __restrict__ out,          // [B]
    int B) {
  int g = blockIdx.x;
  int j = threadIdx.x;
  const float* q = qstar + (size_t)g * 128;
  float v = b4[j];
  for (int k = 0; k < 128; ++k) v = fmaf(q[k], w4[k * 128 + j], v);
  v = fmaf(tvec[g], w4[128 * 128 + j], v);
  v = fmaf(pvec[g], w4[129 * 128 + j], v);
  v = fmaxf(v, 0.0f);
  __shared__ float red[128];
  red[j] = v * w5[j];
  __syncthreads();
  for (int s = 64; s > 0; s >>= 1) {
    if (j < s) red[j] += red[j + s];
    __syncthreads();
  }
  if (j == 0) out[g] = red[0] + b5[0];
}

// ------------------------------- launcher ----------------------------------

extern "C" void kernel_launch(void* const* d_in, const int* in_sizes, int n_in,
                              void* d_out, int out_size, void* d_ws, size_t ws_size,
                              hipStream_t stream) {
  const int IN = 32, H1 = 128, D = 64;
  const int N = in_sizes[0] / IN;   // 200000
  const int E = in_sizes[1] / 6;    // 800000
  const int B = in_sizes[2];        // 4096

  const float* x      = (const float*)d_in[0];
  const float* eattr  = (const float*)d_in[1];
  const float* tvec   = (const float*)d_in[2];
  const float* pvec   = (const float*)d_in[3];
  const float* e1w = (const float*)d_in[4];   const float* e1b = (const float*)d_in[5];
  const float* n1w = (const float*)d_in[6];   const float* n1b = (const float*)d_in[7];
  const float* l1w = (const float*)d_in[8];   const float* l1b = (const float*)d_in[9];
  const float* e2w = (const float*)d_in[10];  const float* e2b = (const float*)d_in[11];
  const float* n2w = (const float*)d_in[12];  const float* n2b = (const float*)d_in[13];
  const float* l2w = (const float*)d_in[14];  const float* l2b = (const float*)d_in[15];
  const float* e3w = (const float*)d_in[16];  const float* e3b = (const float*)d_in[17];
  const float* n3w = (const float*)d_in[18];  const float* n3b = (const float*)d_in[19];
  const float* l3w = (const float*)d_in[20];  const float* l3b = (const float*)d_in[21];
  const float* wih = (const float*)d_in[22];  const float* whh = (const float*)d_in[23];
  const float* bih = (const float*)d_in[24];  const float* bhh = (const float*)d_in[25];
  const float* w4  = (const float*)d_in[26];  const float* b4  = (const float*)d_in[27];
  const float* w5  = (const float*)d_in[28];  const float* b5  = (const float*)d_in[29];
  const int* srcv  = (const int*)d_in[30];          // edge_index row 0
  const int* dstv  = ((const int*)d_in[30]) + E;    // edge_index row 1
  const int* batch = (const int*)d_in[31];
  float* out = (float*)d_out;

  // ----- workspace carving (all offsets 256B aligned) -----
  char* base = (char*)d_ws;
  size_t off = 0;
  auto carve = [&](size_t bytes) {
    void* p = base + off;
    off += (bytes + 255) & ~(size_t)255;
    return p;
  };
  const size_t NB = (size_t)N * 128 * sizeof(float);
  float*  aggr  = (float*)carve(NB);                  // message accumulator
  float*  hA    = (float*)carve(NB);                  // layer buffer A
  float*  hB    = (float*)carve(NB);                  // layer buffer B
  __bf16* Wc    = (__bf16*)carve(2 * 128 * 128 * sizeof(__bf16));
  float*  bc    = (float*)carve(128 * sizeof(float));
  float*  hbufA = (float*)carve((size_t)B * 64 * sizeof(float));
  float*  cbuf  = (float*)carve((size_t)B * 64 * sizeof(float));
  float*  qstar = (float*)carve((size_t)B * 128 * sizeof(float));
  float*  hbufB = (float*)carve((size_t)B * 64 * sizeof(float));
  // contiguous per-step zero region: [emax (B u32)][asum (B f32)][rbuf (B*64 f32)]
  unsigned* emax = (unsigned*)carve((size_t)B * sizeof(unsigned));
  float*    asum = (float*)carve((size_t)B * sizeof(float));
  float*    rbuf = (float*)carve((size_t)B * 64 * sizeof(float));
  float*    evec = (float*)carve((size_t)N * sizeof(float));
  float*    avec = (float*)carve((size_t)N * sizeof(float));
  (void)ws_size; (void)n_in; (void)out_size;

  auto zero = [&](void* p, size_t nElems) {
    zero_f32<<<(unsigned)((nElems + 255) / 256), 256, 0, stream>>>((float*)p, nElems);
  };

  // ----- per-layer edge messages + weight combine -----
  auto edge_phase = [&](const float* xin, const float* ew, const float* eb,
                        const float* nnw, const float* nnb,
                        const float* lw, const float* lb, int dIn, int dOut) {
    zero(aggr, (size_t)N * dIn);
    {
      size_t tot = (size_t)E * dIn;
      msg_aggr<<<(unsigned)((tot + 255) / 256), 256, 0, stream>>>(
          xin, eattr, srcv, dstv, ew, eb, aggr, E, dIn);
    }
    {
      int tot = 2 * dIn * dOut;
      combine_weights<<<(tot + 255) / 256, 256, 0, stream>>>(
          nnw, lw, nnb, lb, Wc, bc, dIn, dOut);
    }
  };
  const int gemmGrid = (N + 63) / 64;  // BM = 64 rows per block

  // Layer 1: x[N,32] -> hA[N,128], elu
  edge_phase(x, e1w, e1b, n1w, n1b, l1w, l1b, IN, H1);
  gemm_wmma<32, 128, true><<<gemmGrid, 256, 0, stream>>>(x, aggr, Wc, bc, hA, N);
  // Layer 2: hA[N,128] -> hB[N,128], elu
  edge_phase(hA, e2w, e2b, n2w, n2b, l2w, l2b, H1, H1);
  gemm_wmma<128, 128, true><<<gemmGrid, 256, 0, stream>>>(hA, aggr, Wc, bc, hB, N);
  // Layer 3: hB[N,128] -> hA[N,64] (h3), no activation
  edge_phase(hB, e3w, e3b, n3w, n3b, l3w, l3b, H1, D);
  gemm_wmma<128, 64, false><<<gemmGrid, 256, 0, stream>>>(hB, aggr, Wc, bc, hA, N);
  const float* h3 = hA;

  // ----- Set2Set (3 steps) -----
  // zero h0, c, q_star (contiguous: hbufA, cbuf, qstar)
  zero(hbufA, (size_t)B * (64 + 64 + 128));
  const int nThreadsBD = B * 64;
  float* hPing = hbufA;
  float* hPong = hbufB;
  for (int step = 0; step < 3; ++step) {
    // zero emax/asum/rbuf (contiguous region of B*(1+1+64) dwords)
    zero(emax, (size_t)B * 66);
    lstm_step<<<(nThreadsBD + 255) / 256, 256, 0, stream>>>(
        qstar, hPing, wih, whh, bih, bhh, hPong, cbuf, B);
    attn_score<<<(N + 255) / 256, 256, 0, stream>>>(h3, hPong, batch, evec, emax, N);
    attn_expsum<<<(N + 255) / 256, 256, 0, stream>>>(evec, emax, batch, avec, asum, N);
    {
      size_t tot = (size_t)N * 64;
      attn_readout<<<(unsigned)((tot + 255) / 256), 256, 0, stream>>>(
          h3, avec, asum, batch, rbuf, N);
    }
    qstar_update<<<(B * 128 + 255) / 256, 256, 0, stream>>>(hPong, rbuf, qstar, B);
    float* tmp = hPing; hPing = hPong; hPong = tmp;
  }

  // ----- head -----
  head_kernel<<<B, 128, 0, stream>>>(qstar, tvec, pvec, w4, b4, w5, b5, out, B);
}
